// ImprovedGRU_9947144257581
// MI455X (gfx1250) — matrix operand order repaired and checked by reading, not compiled, not run on hardware
//
#include <hip/hip_runtime.h>
#include <math.h>

// ---------------------------------------------------------------------------
// Fused 2-layer attention-gated GRU for MI455X (gfx1250, wave32, WMMA).
// One wave owns 16 sequences (WMMA M). All matmuls: v_wmma_f32_16x16x32_f16.
//   - Weight B-fragments live in registers; K<=16 matrices are PACKED in
//     pairs along K (rows 0-15 / 16-31) and selected via A_lo / A_hi.
//   - C operand is inline-0 except z-gate chaining (x@W_z fed as C); biases folded into elementwise stage.
//   - Hidden state bounces through LDS (f16) to re-enter as A-fragments.
// ---------------------------------------------------------------------------

typedef __attribute__((ext_vector_type(16))) _Float16 v16h;
typedef __attribute__((ext_vector_type(8)))  _Float16 v8h;
typedef __attribute__((ext_vector_type(8)))  float    v8f;

#define T_STEPS 64
#define FIN     16
#define H0      32
#define H1      10
#define STILE   16   // sequences per wave tile (= WMMA M)

struct PtrPack { const float* p[29]; };

__device__ __forceinline__ v8f wmma16(v16h a, v16h b, v8f c) {
  // (neg_a, A, neg_b, B, c_mod, C, reuse_a, reuse_b)
  return __builtin_amdgcn_wmma_f32_16x16x32_f16(false, a, false, b, (short)0, c,
                                                false, false);
}

__device__ __forceinline__ v8f zero8f() {
  v8f c;
#pragma unroll
  for (int e = 0; e < 8; ++e) c[e] = 0.0f;
  return c;
}

// Single-matrix B fragment (Kx16 tile, col offset n0) from row-major W[Kd][Nd].
// Lane l: column N = l%16; elems 0-7 K=8h+e, elems 8-15 K=16+8h+(e-8), h=l/16.
__device__ __forceinline__ v16h load_bfrag(const float* W, int Kd, int Nd,
                                           int n0, int lane) {
  const int col = n0 + (lane & 15);
  const int k0  = (lane & 16) >> 1;
  v16h b;
#pragma unroll
  for (int e = 0; e < 16; ++e) {
    const int k = k0 + (e & 7) + ((e & 8) << 1);
    float v = 0.0f;
    if (k < Kd && col < Nd) v = W[k * Nd + col];
    b[e] = (_Float16)v;
  }
  return b;
}

// Packed B fragment: Wlo occupies K rows 0-15, Whi occupies K rows 16-31.
// Use with A_lo (data in K0-15) to get x@Wlo, A_hi (data in K16-31) -> x@Whi.
__device__ __forceinline__ v16h load_bfrag2(const float* Wlo, int Klo,
                                            const float* Whi, int Khi,
                                            int Nd, int n0, int lane) {
  const int col = n0 + (lane & 15);
  const int r0 = (lane & 16) >> 1;
  v16h b;
#pragma unroll
  for (int e = 0; e < 16; ++e) {
    const bool hi = (e & 8) != 0;
    const float* W = hi ? Whi : Wlo;
    const int Kd   = hi ? Khi : Klo;
    const int k = r0 + (e & 7);
    float v = 0.0f;
    if (k < Kd && col < Nd) v = W[k * Nd + col];
    b[e] = (_Float16)v;
  }
  return b;
}

__device__ __forceinline__ float bias_at(const float* b, int Nd, int n0, int lane) {
  const int col = n0 + (lane & 15);
  return (col < Nd) ? b[col] : 0.0f;
}

// Full A-fragment (16x32) from an LDS row of 32 halves.
// Lane l<16: chunks [0,8)+[16,24); lane>=16: [8,16)+[24,32). 16B-aligned.
__device__ __forceinline__ v16h load_afrag_lds(const _Float16* row, int c0) {
  v8h lo = *(const v8h*)(row + c0);
  v8h hi = *(const v8h*)(row + c0 + 16);
  v16h a;
#pragma unroll
  for (int e = 0; e < 8; ++e) { a[e] = lo[e]; a[e + 8] = hi[e]; }
  return a;
}

__device__ __forceinline__ float sigm(float x) { return 1.0f / (1.0f + __expf(-x)); }
__device__ __forceinline__ float tanh_fast(float x) {
  x = fminf(fmaxf(x, -15.0f), 15.0f);
  const float e = __expf(2.0f * x);
  return (e - 1.0f) / (e + 1.0f);
}

__global__ __launch_bounds__(32)
void gru_fused_wmma_kernel(PtrPack P, float* __restrict__ out, int BN) {
  extern __shared__ _Float16 smem[];
  _Float16* ys  = smem;                               // [T+1][16][32] f16 (slot 0 = zeros)
  _Float16* h1b = smem + (T_STEPS + 1) * STILE * H0;  // [16][16]      f16 layer-1 state

  const int lane = threadIdx.x & 31;
  const int lh   = (lane >> 4) & 1;   // lane-half: M rows 0-7 vs 8-15 in C/D
  const int col  = lane & 15;         // N column / A-fragment row (sequence)
  const int c0   = lh ? 8 : 0;        // A-fragment K-chunk base for this lane
  const int seq_base = blockIdx.x * STILE;
  const float* X = P.p[0];

  // Zero the h(-1) slot and the padded layer-1 state (NaN*0 = NaN in WMMA).
  for (int i = lane; i < STILE * H0; i += 32)    ys[i]  = (_Float16)0.0f;
  for (int i = lane; i < STILE * STILE; i += 32) h1b[i] = (_Float16)0.0f;
  __syncthreads();

  const v8f zc = zero8f();  // inline-0 C operand

  // ================= Layer 0 (F_IN=16 -> H0=32) =================
  // Packed x-projection fragments: K0-15 = W_z|W_v rows, K16-31 = W_k|W_h rows.
  v16h Bzk0 = load_bfrag2(P.p[1], FIN, P.p[7],  FIN, H0, 0,  lane);
  v16h Bzk1 = load_bfrag2(P.p[1], FIN, P.p[7],  FIN, H0, 16, lane);
  v16h Bvh0 = load_bfrag2(P.p[9], FIN, P.p[11], FIN, H0, 0,  lane);
  v16h Bvh1 = load_bfrag2(P.p[9], FIN, P.p[11], FIN, H0, 16, lane);
  // Recurrent fragments (K = H0 = 32, full).
  v16h BUz0 = load_bfrag(P.p[3],  H0, H0, 0, lane), BUz1 = load_bfrag(P.p[3],  H0, H0, 16, lane);
  v16h Bq0  = load_bfrag(P.p[5],  H0, H0, 0, lane), Bq1  = load_bfrag(P.p[5],  H0, H0, 16, lane);
  v16h BUh0 = load_bfrag(P.p[13], H0, H0, 0, lane), BUh1 = load_bfrag(P.p[13], H0, H0, 16, lane);

  const float bz0 = bias_at(P.p[2], H0, 0, lane)  + bias_at(P.p[4], H0, 0, lane);
  const float bz1 = bias_at(P.p[2], H0, 16, lane) + bias_at(P.p[4], H0, 16, lane);
  const float bq0 = bias_at(P.p[6],  H0, 0, lane), bq1 = bias_at(P.p[6],  H0, 16, lane);
  const float bk0 = bias_at(P.p[8],  H0, 0, lane), bk1 = bias_at(P.p[8],  H0, 16, lane);
  const float bv0 = bias_at(P.p[10], H0, 0, lane), bv1 = bias_at(P.p[10], H0, 16, lane);
  const float bh0 = bias_at(P.p[12], H0, 0, lane), bh1 = bias_at(P.p[12], H0, 16, lane);
  const float bu0 = bias_at(P.p[14], H0, 0, lane), bu1 = bias_at(P.p[14], H0, 16, lane);

  v8f h0A = zero8f(), h0B = zero8f();        // D-layout state (cols 0-15 / 16-31)
  const float scale0 = 0.1767766952966369f;  // 1/sqrt(32)
  const float* xbase = X + (size_t)(seq_base + col) * (T_STEPS * FIN);

#pragma unroll 1
  for (int t = 0; t < T_STEPS; ++t) {
    // A_x lo/hi: x row chunk placed in K0-15 (lo) or K16-31 (hi).
    const float4* xp = (const float4*)(xbase + t * FIN + c0);
    float4 f0 = xp[0], f1 = xp[1];
    if (t + 1 < T_STEPS) __builtin_prefetch(xbase + (t + 1) * FIN, 0, 1);
    _Float16 xc[8];
    xc[0]=(_Float16)f0.x; xc[1]=(_Float16)f0.y; xc[2]=(_Float16)f0.z; xc[3]=(_Float16)f0.w;
    xc[4]=(_Float16)f1.x; xc[5]=(_Float16)f1.y; xc[6]=(_Float16)f1.z; xc[7]=(_Float16)f1.w;
    v16h AxL, AxH;
#pragma unroll
    for (int e = 0; e < 8; ++e) {
      AxL[e] = xc[e];          AxL[e + 8] = (_Float16)0.0f;
      AxH[e] = (_Float16)0.0f; AxH[e + 8] = xc[e];
    }

    // A_h: previous hidden state (ys slot t; slot 0 is the zero state).
    v16h Ah = load_afrag_lds(ys + (t * STILE + col) * H0, c0);

    v8f prod, zp0, zp1, xv0, xv1, xh0, xh1, uh0, uh1;
    { // ---- column half 0 ----
      v8f txz = wmma16(AxL, Bzk0, zc);
      zp0     = wmma16(Ah,  BUz0, txz);
      v8f tq  = wmma16(Ah,  Bq0,  zc);
      v8f tk  = wmma16(AxH, Bzk0, zc);
#pragma unroll
      for (int r = 0; r < 8; ++r) prod[r] = (tq[r] + bq0) * (tk[r] + bk0);
      xv0 = wmma16(AxL, Bvh0, zc);
      xh0 = wmma16(AxH, Bvh0, zc);
      uh0 = wmma16(Ah,  BUh0, zc);
    }
    { // ---- column half 1 ----
      v8f txz = wmma16(AxL, Bzk1, zc);
      zp1     = wmma16(Ah,  BUz1, txz);
      v8f tq  = wmma16(Ah,  Bq1,  zc);
      v8f tk  = wmma16(AxH, Bzk1, zc);
#pragma unroll
      for (int r = 0; r < 8; ++r) prod[r] += (tq[r] + bq1) * (tk[r] + bk1);
      xv1 = wmma16(AxL, Bvh1, zc);
      xh1 = wmma16(AxH, Bvh1, zc);
      uh1 = wmma16(Ah,  BUh1, zc);
    }

    // alpha = sigmoid(<q,xk>/sqrt(H0)): 16-lane xor-tree reduction over N.
#pragma unroll
    for (int m = 1; m < 16; m <<= 1) {
#pragma unroll
      for (int r = 0; r < 8; ++r) prod[r] += __shfl_xor(prod[r], m, 16);
    }
#pragma unroll
    for (int r = 0; r < 8; ++r) {
      const float alpha = sigm(prod[r] * scale0);
      const float z0 = sigm(zp0[r] + bz0);
      const float z1 = sigm(zp1[r] + bz1);
      const float t0 = tanh_fast((xh0[r] + bh0) + alpha * (xv0[r] + bv0) * (uh0[r] + bu0));
      const float t1 = tanh_fast((xh1[r] + bh1) + alpha * (xv1[r] + bv1) * (uh1[r] + bu1));
      h0A[r] += z0 * (t0 - h0A[r]);
      h0B[r] += z1 * (t1 - h0B[r]);
    }
    // New h (D layout: seq = r+8*lh, hidden = col / 16+col) -> ys slot t+1.
    _Float16* dst = ys + ((t + 1) * STILE) * H0;
#pragma unroll
    for (int r = 0; r < 8; ++r) {
      const int srow = r + 8 * lh;
      dst[srow * H0 + col]      = (_Float16)h0A[r];
      dst[srow * H0 + 16 + col] = (_Float16)h0B[r];
    }
    __syncthreads();
  }

  // ================= Layer 1 (H0=32 -> H1=10, cols padded to 16) ===========
  v16h B1z  = load_bfrag(P.p[15], H0, H1, 0, lane);
  v16h B1k  = load_bfrag(P.p[21], H0, H1, 0, lane);
  v16h B1v  = load_bfrag(P.p[23], H0, H1, 0, lane);
  v16h B1h  = load_bfrag(P.p[25], H0, H1, 0, lane);
  v16h Bzq1 = load_bfrag2(P.p[17], H1, P.p[19], H1, H1, 0, lane); // U_z | W_q packed
  v16h B1u  = load_bfrag(P.p[27], H1, H1, 0, lane);
  const float c1z = bias_at(P.p[16], H1, 0, lane) + bias_at(P.p[18], H1, 0, lane);
  const float c1q = bias_at(P.p[20], H1, 0, lane);
  const float c1k = bias_at(P.p[22], H1, 0, lane);
  const float c1v = bias_at(P.p[24], H1, 0, lane);
  const float c1h = bias_at(P.p[26], H1, 0, lane);
  const float c1u = bias_at(P.p[28], H1, 0, lane);

  v8f h1 = zero8f();
  const float scale1 = 0.31622776601683794f;  // 1/sqrt(10)

#pragma unroll 1
  for (int t = 0; t < T_STEPS; ++t) {
    v16h Ax1 = load_afrag_lds(ys + ((t + 1) * STILE + col) * H0, c0);  // layer-0 out
    // Layer-1 state: only cols 0-15 exist; build lo (K0-15) and hi (K16-31).
    v8h hchunk = *(const v8h*)(h1b + col * STILE + c0);
    v16h AhLo, AhHi;
#pragma unroll
    for (int e = 0; e < 8; ++e) {
      AhLo[e] = hchunk[e];      AhLo[e + 8] = (_Float16)0.0f;
      AhHi[e] = (_Float16)0.0f; AhHi[e + 8] = hchunk[e];
    }

    v8f txz = wmma16(Ax1,  B1z,  zc);
    v8f zp  = wmma16(AhLo, Bzq1, txz);  // + h@U_z (K0-9 rows)
    v8f q   = wmma16(AhHi, Bzq1, zc);   //   h@W_q (K16-25 rows)
    v8f xk  = wmma16(Ax1,  B1k,  zc);
    v8f xv  = wmma16(Ax1,  B1v,  zc);
    v8f xh  = wmma16(Ax1,  B1h,  zc);
    v8f uh  = wmma16(AhLo, B1u,  zc);

    v8f prod;
#pragma unroll
    for (int r = 0; r < 8; ++r) prod[r] = (q[r] + c1q) * (xk[r] + c1k);
#pragma unroll
    for (int m = 1; m < 16; m <<= 1) {
#pragma unroll
      for (int r = 0; r < 8; ++r) prod[r] += __shfl_xor(prod[r], m, 16);
    }
    // cols >= 10 have zero q/xk/bias, so they contribute 0 to the reduction,
    // but their lanes still get garbage alpha -> harmless (state never stored).
#pragma unroll
    for (int r = 0; r < 8; ++r) {
      const float alpha = sigm(prod[r] * scale1);
      const float zv = sigm(zp[r] + c1z);
      const float ht = tanh_fast((xh[r] + c1h) + alpha * (xv[r] + c1v) * (uh[r] + c1u));
      h1[r] += zv * (ht - h1[r]);
    }
    __syncthreads();
    if (col < H1) {
#pragma unroll
      for (int r = 0; r < 8; ++r) h1b[(r + 8 * lh) * STILE + col] = (_Float16)h1[r];
    }
    __syncthreads();
  }

  // Final output: h1 at t=T-1, [BN, 10] fp32.
  if (col < H1) {
#pragma unroll
    for (int r = 0; r < 8; ++r) {
      const int s = seq_base + r + 8 * lh;
      if (s < BN) out[(size_t)s * H1 + col] = h1[r];
    }
  }
}

extern "C" void kernel_launch(void* const* d_in, const int* in_sizes, int n_in,
                              void* d_out, int out_size, void* d_ws, size_t ws_size,
                              hipStream_t stream) {
  (void)d_ws; (void)ws_size; (void)out_size;
  PtrPack pk;
  for (int i = 0; i < 29; ++i)
    pk.p[i] = (i < n_in) ? (const float*)d_in[i] : nullptr;

  const int BN = in_sizes[0] / (T_STEPS * FIN);   // 16000 sequences
  const int blocks = (BN + STILE - 1) / STILE;    // 1000 tiles, 1 wave each
  const size_t lds_bytes =
      (size_t)((T_STEPS + 1) * STILE * H0 + STILE * STILE) * sizeof(_Float16);

  gru_fused_wmma_kernel<<<blocks, 32, lds_bytes, stream>>>(pk, (float*)d_out, BN);
}
